// BatchTreeEncoder_90460601189009
// MI455X (gfx1250) — compile-verified, matrix operand order checked by reading, not run on hardware
//
#include <hip/hip_runtime.h>
#include <hip/hip_bf16.h>

// ---------------------------------------------------------------------------
// BatchTreeEncoder for MI455X (gfx1250, wave32, WMMA)
// BS=256, K=3, DEPTH=4, E=1024, N_NODES=121
// Strategy: bf16 activations/weights, v_wmma_f32_16x16x32_bf16 for all GEMMs,
// fp32 accumulate + fp32 pointwise GRU/attention math.
// ---------------------------------------------------------------------------

#define E 1024
#define BSZ 256
#define NNODES 121

typedef __bf16 v16bf __attribute__((ext_vector_type(16)));
typedef float  v8f   __attribute__((ext_vector_type(8)));

union Frag16 { v16bf v; uint4 q[2]; };

// Load one WMMA 16-bit operand fragment (A row-layout / mirrored B layout):
// lane L holds tile row (L&15); K-halves: lanes 0-15 get K=[0..8)+[16..24),
// lanes 16-31 get K=[8..16)+[24..32). Two 16-byte chunks per lane.
__device__ __forceinline__ Frag16 load_frag(const __hip_bfloat16* __restrict__ rowp,
                                            int k0, int kl) {
  Frag16 f;
  f.q[0] = *reinterpret_cast<const uint4*>(rowp + k0 + kl);
  f.q[1] = *reinterpret_cast<const uint4*>(rowp + k0 + 16 + kl);
  return f;
}

__device__ __forceinline__ v8f wmma_bf16(const Frag16& a, const Frag16& b, v8f c) {
  return __builtin_amdgcn_wmma_f32_16x16x32_bf16(
      /*neg_a=*/false, a.v, /*neg_b=*/false, b.v,
      /*c_mod=*/(short)0, c, /*reuse_a=*/false, /*reuse_b=*/false);
}

__device__ __forceinline__ float sigmoidf_fast(float x) {
  return 1.0f / (1.0f + __expf(-x));
}

// ---------------------------------------------------------------------------
// Prep kernels
// ---------------------------------------------------------------------------
__global__ void k_f32_to_bf16(const float* __restrict__ src,
                              __hip_bfloat16* __restrict__ dst, int n) {
  int i = blockIdx.x * blockDim.x + threadIdx.x;
  if (i < n) dst[i] = __float2bfloat16(src[i]);
}

// sent_weight (E x F, contracted index e first) -> transposed bf16 (F x E)
__global__ void k_transpose_bf16(const float* __restrict__ src,
                                 __hip_bfloat16* __restrict__ dst) {
  int i = blockIdx.x * blockDim.x + threadIdx.x;   // over E*E
  int f = i >> 10;
  int e = i & (E - 1);
  dst[(size_t)f * E + e] = __float2bfloat16(src[(size_t)e * E + f]);
}

// x[bn, :] = bf16(emb[tokens[b, tokOff+n], :])
__global__ void k_gather_embed(const float* __restrict__ emb,
                               const int* __restrict__ tokens,
                               int tokOff, int n_l,
                               __hip_bfloat16* __restrict__ Xb) {
  int bn = blockIdx.x;
  int b = bn / n_l;
  int n = bn - b * n_l;
  int tok = tokens[b * NNODES + tokOff + n];
  const float* src = emb + (size_t)tok * E;
  __hip_bfloat16* dst = Xb + (size_t)bn * E;
  for (int e = threadIdx.x; e < E; e += blockDim.x)
    dst[e] = __float2bfloat16(src[e]);
}

// ---------------------------------------------------------------------------
// Attention scores: s[row] = tanh( tanh(child @ sentW + sbias) . ctx )
// One wave per 16-row tile; sweeps all 64 column tiles of the 1024-wide
// projection, fusing tanh + dot(context) so U is never written to memory.
// ---------------------------------------------------------------------------
__global__ void __launch_bounds__(256)
k_attn_score(const __hip_bfloat16* __restrict__ ChildB,
             const __hip_bfloat16* __restrict__ SWT,   // (F=1024 rows, E contig)
             const float* __restrict__ sbias,
             const float* __restrict__ ctx,
             float* __restrict__ s, int Mrows) {
  int wave = threadIdx.x >> 5;
  int lane = threadIdx.x & 31;
  int rowTile = blockIdx.x * 8 + wave;
  if (rowTile * 16 >= Mrows) return;

  int lr = lane & 15;     // tile row for A, tile col for B, output col for C
  int hi = lane >> 4;
  int kl = hi * 8;

  const __hip_bfloat16* arow = ChildB + ((size_t)rowTile * 16 + lr) * E;

  float partial[8];
#pragma unroll
  for (int j = 0; j < 8; ++j) partial[j] = 0.0f;

  for (int ct = 0; ct < 64; ++ct) {
    int n0 = ct * 16;
    const __hip_bfloat16* brow = SWT + (size_t)(n0 + lr) * E;
    v8f acc = {0.f, 0.f, 0.f, 0.f, 0.f, 0.f, 0.f, 0.f};
    for (int k0 = 0; k0 < E; k0 += 32) {
      Frag16 a = load_frag(arow, k0, kl);
      Frag16 b = load_frag(brow, k0, kl);
      acc = wmma_bf16(a, b, acc);
    }
    int n = n0 + lr;
    float cb = ctx[n];
    float sb = sbias[n];
#pragma unroll
    for (int j = 0; j < 8; ++j) {
      float u = tanhf(acc[j] + sb);
      partial[j] += u * cb;
    }
  }

  // reduce over the 16 lanes that share each output row
#pragma unroll
  for (int j = 0; j < 8; ++j) {
    float v = partial[j];
    v += __shfl_xor(v, 1);
    v += __shfl_xor(v, 2);
    v += __shfl_xor(v, 4);
    v += __shfl_xor(v, 8);
    if (lr == 0) s[rowTile * 16 + hi * 8 + j] = tanhf(v);
  }
}

// ---------------------------------------------------------------------------
// softmax over K=3 children + weighted sum -> hsum (bf16)
// child rows are (b*n_l + n)*3 + k
// ---------------------------------------------------------------------------
__global__ void k_softmax_wsum(const float* __restrict__ s,
                               const float* __restrict__ ChildF,
                               __hip_bfloat16* __restrict__ Hsb) {
  int bn = blockIdx.x;
  float s0 = s[bn * 3 + 0], s1 = s[bn * 3 + 1], s2 = s[bn * 3 + 2];
  float mx = fmaxf(s0, fmaxf(s1, s2));
  float w0 = __expf(s0 - mx), w1 = __expf(s1 - mx), w2 = __expf(s2 - mx);
  float inv = 1.0f / (w0 + w1 + w2);
  w0 *= inv; w1 *= inv; w2 *= inv;
  const float* c0 = ChildF + (size_t)(bn * 3 + 0) * E;
  const float* c1 = ChildF + (size_t)(bn * 3 + 1) * E;
  const float* c2 = ChildF + (size_t)(bn * 3 + 2) * E;
  __hip_bfloat16* dst = Hsb + (size_t)bn * E;
  for (int e = threadIdx.x; e < E; e += blockDim.x)
    dst[e] = __float2bfloat16(w0 * c0[e] + w1 * c1[e] + w2 * c2[e]);
}

// ---------------------------------------------------------------------------
// Fused GRU: per 16x16 output tile carry 6 WMMA accumulators
// (gates r/z/n for both x@Wi^T and hsum@Wh^T), then pointwise GRU.
// Writes h as fp32 (for weighted sum / final max) and bf16 (next level's A).
// grid = (Mrows/16, 8), block = 256 (8 waves -> 8 column tiles per block)
// ---------------------------------------------------------------------------
__global__ void __launch_bounds__(256)
k_gru_gemm(const __hip_bfloat16* __restrict__ Xb,
           const __hip_bfloat16* __restrict__ Hsb,
           const __hip_bfloat16* __restrict__ WiB,   // (3072 rows, E contig)
           const __hip_bfloat16* __restrict__ WhB,   // (3072 rows, E contig)
           const float* __restrict__ bi,
           const float* __restrict__ bh,
           float* __restrict__ Hf,
           __hip_bfloat16* __restrict__ Hb,
           int Mrows, int has_h) {
  int wave = threadIdx.x >> 5;
  int lane = threadIdx.x & 31;
  int rowTile = blockIdx.x;
  int colTile = blockIdx.y * 8 + wave;
  int e0 = colTile * 16;

  int lr = lane & 15;
  int hi = lane >> 4;
  int kl = hi * 8;

  const __hip_bfloat16* ax = Xb + ((size_t)rowTile * 16 + lr) * E;
  const __hip_bfloat16* ah = Hsb + ((size_t)rowTile * 16 + lr) * E;
  const __hip_bfloat16* wir = WiB + (size_t)(0 * E + e0 + lr) * E;
  const __hip_bfloat16* wiz = WiB + (size_t)(1 * E + e0 + lr) * E;
  const __hip_bfloat16* win = WiB + (size_t)(2 * E + e0 + lr) * E;
  const __hip_bfloat16* whr = WhB + (size_t)(0 * E + e0 + lr) * E;
  const __hip_bfloat16* whz = WhB + (size_t)(1 * E + e0 + lr) * E;
  const __hip_bfloat16* whn = WhB + (size_t)(2 * E + e0 + lr) * E;

  v8f axr = {0.f,0.f,0.f,0.f,0.f,0.f,0.f,0.f};
  v8f axz = axr, axn = axr, ahr = axr, ahz = axr, ahn = axr;

  for (int k0 = 0; k0 < E; k0 += 32) {
    __builtin_prefetch(ax + k0 + 128, 0, 0);   // stream A ahead (global_prefetch)
    Frag16 a = load_frag(ax, k0, kl);
    Frag16 b;
    b = load_frag(wir, k0, kl); axr = wmma_bf16(a, b, axr);
    b = load_frag(wiz, k0, kl); axz = wmma_bf16(a, b, axz);
    b = load_frag(win, k0, kl); axn = wmma_bf16(a, b, axn);
    if (has_h) {
      Frag16 a2 = load_frag(ah, k0, kl);
      b = load_frag(whr, k0, kl); ahr = wmma_bf16(a2, b, ahr);
      b = load_frag(whz, k0, kl); ahz = wmma_bf16(a2, b, ahz);
      b = load_frag(whn, k0, kl); ahn = wmma_bf16(a2, b, ahn);
    }
  }

  // pointwise GRU on the 16x16 tile (C layout: vgpr j, lanes 0-15 -> row j,
  // lanes 16-31 -> row j+8; column = lane&15)
  int e = e0 + lr;
  float bir = bi[e], biz = bi[E + e], bin = bi[2 * E + e];
  float bhr = bh[e], bhz = bh[E + e], bhn = bh[2 * E + e];
#pragma unroll
  for (int j = 0; j < 8; ++j) {
    int row = rowTile * 16 + hi * 8 + j;
    float gxr = axr[j] + bir;
    float gxz = axz[j] + biz;
    float gxn = axn[j] + bin;
    float ghr = bhr, ghz = bhz, ghn = bhn, hs = 0.0f;
    if (has_h) {
      ghr += ahr[j]; ghz += ahz[j]; ghn += ahn[j];
      hs = __bfloat162float(Hsb[(size_t)row * E + e]);
    }
    float r  = sigmoidf_fast(gxr + ghr);
    float z  = sigmoidf_fast(gxz + ghz);
    float nn = tanhf(gxn + r * ghn);
    float h  = (1.0f - z) * nn + z * hs;
    Hf[(size_t)row * E + e] = h;
    Hb[(size_t)row * E + e] = __float2bfloat16(h);
  }
}

// ---------------------------------------------------------------------------
// out[b, e] = max over all 121 node states
// ---------------------------------------------------------------------------
__global__ void k_final_max(const float* __restrict__ Hf, float* __restrict__ out) {
  int idx = blockIdx.x * blockDim.x + threadIdx.x;   // 0..BSZ*E-1
  int b = idx >> 10;
  int e = idx & (E - 1);
  const int rb[5] = {30720, 29952, 27648, 20736, 0};
  const int nl[5] = {1, 3, 9, 27, 81};
  float m = -3.4e38f;
#pragma unroll
  for (int l = 0; l < 5; ++l) {
    int base = rb[l] + b * nl[l];
    for (int n = 0; n < nl[l]; ++n)
      m = fmaxf(m, Hf[(size_t)(base + n) * E + e]);
  }
  out[idx] = m;
}

// ---------------------------------------------------------------------------
// Host side
// ---------------------------------------------------------------------------
extern "C" void kernel_launch(void* const* d_in, const int* in_sizes, int n_in,
                              void* d_out, int out_size, void* d_ws, size_t ws_size,
                              hipStream_t stream) {
  (void)in_sizes; (void)n_in; (void)out_size; (void)ws_size;

  const int*   tokens = (const int*)d_in[0];
  // d_in[1] is scalar bs (==256), ignored
  const float* emb    = (const float*)d_in[2];
  const float* Wi     = (const float*)d_in[3];
  const float* Wh     = (const float*)d_in[4];
  const float* bi     = (const float*)d_in[5];
  const float* bh     = (const float*)d_in[6];
  const float* sentW  = (const float*)d_in[7];
  const float* sbias  = (const float*)d_in[8];
  const float* ctx    = (const float*)d_in[9];
  float* out = (float*)d_out;

  char* ws = (char*)d_ws;
  size_t off = 0;
  auto take = [&](size_t bytes) -> char* {
    char* p = ws + off;
    off += (bytes + 255) & ~(size_t)255;
    return p;
  };
  const int maxRows = BSZ * 81;                         // 20736
  __hip_bfloat16* wiB  = (__hip_bfloat16*)take((size_t)3 * E * E * 2);
  __hip_bfloat16* whB  = (__hip_bfloat16*)take((size_t)3 * E * E * 2);
  __hip_bfloat16* swtB = (__hip_bfloat16*)take((size_t)E * E * 2);
  __hip_bfloat16* Xb   = (__hip_bfloat16*)take((size_t)maxRows * E * 2);
  __hip_bfloat16* Hsb  = (__hip_bfloat16*)take((size_t)maxRows * E * 2);
  __hip_bfloat16* Hb   = (__hip_bfloat16*)take((size_t)maxRows * E * 2);
  float*          Hf   = (float*)take((size_t)30976 * E * 4);
  float*          sbuf = (float*)take((size_t)maxRows * 4);

  // one-time weight prep (deterministic each call)
  k_f32_to_bf16<<<(3 * E * E + 255) / 256, 256, 0, stream>>>(Wi, wiB, 3 * E * E);
  k_f32_to_bf16<<<(3 * E * E + 255) / 256, 256, 0, stream>>>(Wh, whB, 3 * E * E);
  k_transpose_bf16<<<(E * E + 255) / 256, 256, 0, stream>>>(sentW, swtB);

  const int nl[5]     = {1, 3, 9, 27, 81};
  const int tokOff[5] = {0, 1, 4, 13, 40};
  const int rb[5]     = {30720, 29952, 27648, 20736, 0};   // row bases in Hf

  // level 4 (leaves): hsum = 0
  {
    int n_l = 81, M = BSZ * n_l;
    k_gather_embed<<<M, 256, 0, stream>>>(emb, tokens, tokOff[4], n_l, Xb);
    k_gru_gemm<<<dim3(M / 16, 8), 256, 0, stream>>>(
        Xb, Hsb, wiB, whB, bi, bh, Hf + (size_t)rb[4] * E, Hb, M, 0);
  }

  for (int l = 3; l >= 0; --l) {
    int n_l = nl[l];
    int Mc  = BSZ * nl[l + 1];   // child rows (previous level)
    int M   = BSZ * n_l;
    k_attn_score<<<Mc / 128, 256, 0, stream>>>(Hb, swtB, sbias, ctx, sbuf, Mc);
    k_softmax_wsum<<<M, 256, 0, stream>>>(sbuf, Hf + (size_t)rb[l + 1] * E, Hsb);
    k_gather_embed<<<M, 256, 0, stream>>>(emb, tokens, tokOff[l], n_l, Xb);
    k_gru_gemm<<<dim3(M / 16, 8), 256, 0, stream>>>(
        Xb, Hsb, wiB, whB, bi, bh, Hf + (size_t)rb[l] * E, Hb, M, 1);
  }

  k_final_max<<<(BSZ * E) / 256, 256, 0, stream>>>(Hf, out);
}
